// GT_31327491457434
// MI455X (gfx1250) — matrix operand order repaired and checked by reading, not compiled, not run on hardware
//
#include <hip/hip_runtime.h>
#include <hip/hip_bf16.h>
#include <math.h>

// ---------------------------------------------------------------------------
// Graph-transformer forward for MI455X (gfx1250, wave32, WMMA).
// B=1, N=50000 nodes, E=800000 edges, D=128, H=8, HD=16, L=3, OUT=10.
// GEMMs: v_wmma_f32_16x16x32_f16, weights pre-packed into fragment order so
// each lane's B fragment is one 32-byte contiguous load; GEMM templated on
// (K,N) so all fragment addresses fold into instruction immediates.
// ---------------------------------------------------------------------------

typedef __attribute__((ext_vector_type(16))) _Float16 v16h;
typedef __attribute__((ext_vector_type(8)))  float    v8f;

#define DIM 128

// ---------------------------------------------------------------------------
// Embedding gather: X[n, :] = emb[idx[n], :]
// ---------------------------------------------------------------------------
__global__ __launch_bounds__(256)
void embed_kernel(const int* __restrict__ idx, const float* __restrict__ emb,
                  float* __restrict__ X, int M) {
  int i = blockIdx.x * blockDim.x + threadIdx.x;
  if (i >= M * DIM) return;
  int n = i >> 7, d = i & (DIM - 1);
  X[i] = emb[(size_t)idx[n] * DIM + d];
}

// ---------------------------------------------------------------------------
// Weight pack: W[K,N] f32 -> P f16 in WMMA B-fragment order.
// Fragment (kt, ntile, grp): lane lr holds 16 contiguous halfs (32 bytes):
//   P[(((kt*nTiles + ntile)*2 + grp)*16 + lr)*16 + j]
//     = (f16) W[kt*32 + 8*grp + (j&7) + 16*(j>>3)][ntile*16 + lr]
// ---------------------------------------------------------------------------
__global__ __launch_bounds__(256)
void pack_w(const float* __restrict__ W, _Float16* __restrict__ P, int K, int N) {
  int i = blockIdx.x * blockDim.x + threadIdx.x;
  if (i >= K * N) return;
  int j     = i & 15;
  int lr    = (i >> 4) & 15;
  int grp   = (i >> 8) & 1;
  int rest  = i >> 9;
  int nTiles = N >> 4;
  int ntile = rest % nTiles;
  int kt    = rest / nTiles;
  P[i] = (_Float16)W[(size_t)(kt * 32 + 8 * grp + (j & 7) + 16 * (j >> 3)) * N + (ntile * 16 + lr)];
}

// ---------------------------------------------------------------------------
// WMMA GEMM: C[M,N] = act( A[M,K] @ W[K,N] + bias[N] ).
// A fp32 (b128 loads, f16 convert in regs); W pre-packed f16 (32B/lane frag).
// Compile-time K,N: fully unrolled k-loop, immediate-offset addressing.
// Block = 256 threads = 8 waves; wave computes a 16x128 output tile.
// Requires M%16==0.
// ---------------------------------------------------------------------------
template <int K, int N, int RELU>
__global__ __launch_bounds__(256)
void gemm_wmma(const float* __restrict__ A, const _Float16* __restrict__ P,
               const float* __restrict__ bias, float* __restrict__ C, int M) {
  const int lane = threadIdx.x & 31;
  const int wave = threadIdx.x >> 5;
  const int rt   = blockIdx.x * 8 + wave;     // 16-row tile index
  if (rt * 16 >= M) return;
  const int m0   = rt * 16;
  const int n0   = blockIdx.y * 128;
  const int lr   = lane & 15;                 // row (A) / col (B,C) within tile
  const int grp  = lane >> 4;                 // half-wave group
  constexpr int nTiles   = N / 16;
  constexpr int fragB    = 2 * 16 * 32;       // bytes per (kt,ntile) fragment pair
  constexpr int ktStride = nTiles * fragB;    // bytes per k-step

  v8f acc[8] = {};

  const float* Arow  = A + (size_t)(m0 + lr) * K + 8 * grp;
  const char*  Bbase = (const char*)P +
      (size_t)((((n0 >> 4) * 2 + grp) * 16 + lr)) * 32;

#pragma unroll
  for (int kt = 0; kt < K / 32; ++kt) {
    // A fragment (16x32 f16): K = kt*32 + {0..7}+8g and kt*32 + {16..23}+8g
    const float4* p0 = (const float4*)(Arow + kt * 32);
    const float4* p1 = (const float4*)(Arow + kt * 32 + 16);
    float4 x0 = p0[0], x1 = p0[1], x2 = p1[0], x3 = p1[1];
    v16h a;
    a[0]  = (_Float16)x0.x; a[1]  = (_Float16)x0.y;
    a[2]  = (_Float16)x0.z; a[3]  = (_Float16)x0.w;
    a[4]  = (_Float16)x1.x; a[5]  = (_Float16)x1.y;
    a[6]  = (_Float16)x1.z; a[7]  = (_Float16)x1.w;
    a[8]  = (_Float16)x2.x; a[9]  = (_Float16)x2.y;
    a[10] = (_Float16)x2.z; a[11] = (_Float16)x2.w;
    a[12] = (_Float16)x3.x; a[13] = (_Float16)x3.y;
    a[14] = (_Float16)x3.z; a[15] = (_Float16)x3.w;

#pragma unroll
    for (int nt = 0; nt < 8; ++nt) {
      // one 32-byte contiguous load per lane, immediate offset
      v16h b = *(const v16h*)(Bbase + kt * ktStride + nt * fragB);
      acc[nt] = __builtin_amdgcn_wmma_f32_16x16x32_f16(
          false, a, false, b, (short)0, acc[nt], false, false);
    }
  }

  // epilogue: bias (+ optional ReLU), store. Row = m0 + r + 8*grp.
#pragma unroll
  for (int nt = 0; nt < 8; ++nt) {
    const int n  = n0 + nt * 16 + lr;
    const float bv = bias[n];
    float* Crow = C + (size_t)(m0 + 8 * grp) * N + n;
#pragma unroll
    for (int r = 0; r < 8; ++r) {
      float val = acc[nt][r] + bv;
      if (RELU) val = fmaxf(val, 0.f);
      Crow[r * N] = val;
    }
  }
}

// ---------------------------------------------------------------------------
// Edge attention: one lane per (edge, head).
// score = exp(clip(<k[src,h], q[dst,h]>/4, -5, 5))
// wV[dst,h,:] += v[src,h,:] * score ; Z[dst,h] += score  (f32 atomics, L2)
// ---------------------------------------------------------------------------
__global__ __launch_bounds__(256)
void edge_attn(const int* __restrict__ ei, const float* __restrict__ Q,
               const float* __restrict__ Kt, const float* __restrict__ V,
               float* __restrict__ WV, float* __restrict__ Z, int E) {
  int idx = blockIdx.x * blockDim.x + threadIdx.x;
  int e = idx >> 3, h = idx & 7;
  if (e >= E) return;
  int s = ei[e];        // edge_index[0] : source
  int d = ei[E + e];    // edge_index[1] : destination
  const float4* qp = (const float4*)(Q  + (size_t)d * DIM + h * 16);
  const float4* kp = (const float4*)(Kt + (size_t)s * DIM + h * 16);
  float dot = 0.f;
#pragma unroll
  for (int i = 0; i < 4; ++i) {
    float4 qv = qp[i], kv = kp[i];
    dot += qv.x * kv.x + qv.y * kv.y + qv.z * kv.z + qv.w * kv.w;
  }
  dot = fminf(fmaxf(dot * 0.25f, -5.f), 5.f);   // /sqrt(HD=16), clip
  float sc = __expf(dot);
  const float* vp = V + (size_t)s * DIM + h * 16;
  float* wp = WV + (size_t)d * DIM + h * 16;
#pragma unroll
  for (int i = 0; i < 16; ++i) atomicAdd(wp + i, vp[i] * sc);
  atomicAdd(Z + (size_t)d * 8 + h, sc);
}

// attn = wV / (Z + 1e-6), in place
__global__ __launch_bounds__(256)
void attn_norm(float* __restrict__ WV, const float* __restrict__ Z, int total) {
  int i = blockIdx.x * blockDim.x + threadIdx.x;
  if (i >= total) return;
  int node = i >> 7, h = (i >> 4) & 7;
  WV[i] = WV[i] / (Z[node * 8 + h] + 1e-6f);
}

// ---------------------------------------------------------------------------
// X = LayerNorm(X + Y) * g + b, one wave per 128-wide row (in place).
// ---------------------------------------------------------------------------
__global__ __launch_bounds__(256)
void add_ln(float* __restrict__ X, const float* __restrict__ Y,
            const float* __restrict__ g, const float* __restrict__ b, int M) {
  int lane = threadIdx.x & 31;
  int row  = blockIdx.x * 8 + (threadIdx.x >> 5);
  if (row >= M) return;
  float* xr = X + (size_t)row * DIM;
  const float* yr = Y + (size_t)row * DIM;
  float v[4];
  float s = 0.f;
#pragma unroll
  for (int i = 0; i < 4; ++i) {
    v[i] = xr[lane + 32 * i] + yr[lane + 32 * i];
    s += v[i];
  }
#pragma unroll
  for (int off = 16; off; off >>= 1) s += __shfl_xor(s, off, 32);
  float mean = s * (1.f / DIM);
  float var = 0.f;
#pragma unroll
  for (int i = 0; i < 4; ++i) { float t = v[i] - mean; var += t * t; }
#pragma unroll
  for (int off = 16; off; off >>= 1) var += __shfl_xor(var, off, 32);
  float inv = rsqrtf(var * (1.f / DIM) + 1e-5f);
#pragma unroll
  for (int i = 0; i < 4; ++i) {
    int c = lane + 32 * i;
    xr[c] = (v[i] - mean) * inv * g[c] + b[c];
  }
}

// column partial sums -> atomic into acc[128]
__global__ __launch_bounds__(128)
void colsum(const float* __restrict__ X, float* __restrict__ acc, int M) {
  int d = threadIdx.x;
  float s = 0.f;
  for (int r = blockIdx.x; r < M; r += gridDim.x) s += X[(size_t)r * DIM + d];
  atomicAdd(acc + d, s);
}

// readout MLP: mean -> 64 (relu) -> 32 (relu) -> 10
__global__ __launch_bounds__(128)
void readout(const float* __restrict__ acc,
             const float* __restrict__ W0, const float* __restrict__ b0,
             const float* __restrict__ W1, const float* __restrict__ b1,
             const float* __restrict__ W2, const float* __restrict__ b2,
             float* __restrict__ out, float invM) {
  __shared__ float m[128], h0[64], h1[32];
  int t = threadIdx.x;
  m[t] = acc[t] * invM;
  __syncthreads();
  if (t < 64) {
    float s = b0[t];
    for (int i = 0; i < 128; ++i) s += m[i] * W0[i * 64 + t];
    h0[t] = fmaxf(s, 0.f);
  }
  __syncthreads();
  if (t < 32) {
    float s = b1[t];
    for (int i = 0; i < 64; ++i) s += h0[i] * W1[i * 32 + t];
    h1[t] = fmaxf(s, 0.f);
  }
  __syncthreads();
  if (t < 10) {
    float s = b2[t];
    for (int i = 0; i < 32; ++i) s += h1[i] * W2[i * 10 + t];
    out[t] = s;
  }
}

// ---------------------------------------------------------------------------
extern "C" void kernel_launch(void* const* d_in, const int* in_sizes, int n_in,
                              void* d_out, int out_size, void* d_ws, size_t ws_size,
                              hipStream_t stream) {
  (void)n_in; (void)out_size; (void)ws_size;
  const int M = in_sizes[0];            // B*N = 50000
  const int E = in_sizes[1] / 2;        // 800000
  const int L = 3;

  const int*   x_idx = (const int*)d_in[0];
  const int*   eidx  = (const int*)d_in[1];
  const float* emb   = (const float*)d_in[2];
  const float* Wq = (const float*)d_in[3],  *bq = (const float*)d_in[4];
  const float* Wk = (const float*)d_in[5],  *bk = (const float*)d_in[6];
  const float* Wv = (const float*)d_in[7],  *bv = (const float*)d_in[8];
  const float* Wo = (const float*)d_in[9],  *bo = (const float*)d_in[10];
  const float* g1 = (const float*)d_in[11], *be1 = (const float*)d_in[12];
  const float* Wf1 = (const float*)d_in[13], *bf1 = (const float*)d_in[14];
  const float* Wf2 = (const float*)d_in[15], *bf2 = (const float*)d_in[16];
  const float* g2 = (const float*)d_in[17], *be2 = (const float*)d_in[18];
  const float* mW0 = (const float*)d_in[19], *mb0 = (const float*)d_in[20];
  const float* mW1 = (const float*)d_in[21], *mb1 = (const float*)d_in[22];
  const float* mW2 = (const float*)d_in[23], *mb2 = (const float*)d_in[24];
  float* out = (float*)d_out;

  // workspace layout (floats). Q|K region doubles as the FFN hidden buffer
  // T[M,256] later (Q/K dead by then).
  const size_t ND = (size_t)M * DIM;
  const size_t NH = (size_t)M * 8;
  float* ws  = (float*)d_ws;
  float* X   = ws;
  float* Y   = ws + ND;
  float* Q   = ws + 2 * ND;
  float* Kb  = ws + 3 * ND;
  float* T   = Q;                        // [M, 256]
  float* V   = ws + 4 * ND;              // also reused as y2
  float* WV  = ws + 5 * ND;
  float* Z   = ws + 6 * ND;              // [M, 8] (contiguous after WV)
  float* ACC = ws + 6 * ND + NH;         // [128]
  // packed f16 weights (32B-aligned: offset divisible by 8 floats)
  _Float16* PK = (_Float16*)(ws + 6 * ND + NH + 256);
  const size_t szQ = (size_t)DIM * DIM;          // per-matrix f16 elem counts
  const size_t szF = (size_t)DIM * 2 * DIM;
  _Float16* pWq = PK;                      // [L][128*128]
  _Float16* pWk = PK + L * szQ;
  _Float16* pWv = PK + 2 * L * szQ;
  _Float16* pWo = PK + 3 * L * szQ;
  _Float16* pWf1 = PK + 4 * L * szQ;       // [L][128*256]
  _Float16* pWf2 = pWf1 + L * szF;         // [L][256*128]

  const int rowTiles = M / 16;                       // 3125
  const dim3 blk(256);
  const dim3 gGemm128((rowTiles + 7) / 8, 1);        // N=128: one col-block
  const dim3 gGemm256((rowTiles + 7) / 8, 2);        // N=256
  const int gEmbed = (M * DIM + 255) / 256;
  const int gEdge  = (E * 8 + 255) / 256;
  const int gNorm  = (int)((ND + 255) / 256);
  const int gLN    = (M + 7) / 8;
  const int gPackQ = (int)((szQ + 255) / 256);
  const int gPackF = (int)((szF + 255) / 256);

  // pack all weights to f16 fragment order (tiny; L2-resident thereafter)
  for (int l = 0; l < L; ++l) {
    pack_w<<<gPackQ, blk, 0, stream>>>(Wq + l * szQ, pWq + l * szQ, DIM, DIM);
    pack_w<<<gPackQ, blk, 0, stream>>>(Wk + l * szQ, pWk + l * szQ, DIM, DIM);
    pack_w<<<gPackQ, blk, 0, stream>>>(Wv + l * szQ, pWv + l * szQ, DIM, DIM);
    pack_w<<<gPackQ, blk, 0, stream>>>(Wo + l * szQ, pWo + l * szQ, DIM, DIM);
    pack_w<<<gPackF, blk, 0, stream>>>(Wf1 + l * szF, pWf1 + l * szF, DIM, 2 * DIM);
    pack_w<<<gPackF, blk, 0, stream>>>(Wf2 + l * szF, pWf2 + l * szF, 2 * DIM, DIM);
  }

  // node embedding
  embed_kernel<<<gEmbed, blk, 0, stream>>>(x_idx, emb, X, M);

  for (int l = 0; l < L; ++l) {
    // Q, K, V projections
    gemm_wmma<128, 128, 0><<<gGemm128, blk, 0, stream>>>(X, pWq + l * szQ, bq + l * DIM, Q,  M);
    gemm_wmma<128, 128, 0><<<gGemm128, blk, 0, stream>>>(X, pWk + l * szQ, bk + l * DIM, Kb, M);
    gemm_wmma<128, 128, 0><<<gGemm128, blk, 0, stream>>>(X, pWv + l * szQ, bv + l * DIM, V,  M);

    // edge attention (wV, Z zeroed; they are contiguous)
    hipMemsetAsync(WV, 0, (ND + NH) * sizeof(float), stream);
    edge_attn<<<gEdge, blk, 0, stream>>>(eidx, Q, Kb, V, WV, Z, E);
    attn_norm<<<gNorm, blk, 0, stream>>>(WV, Z, (int)ND);

    // output projection, residual + LN1 (in place into X)
    gemm_wmma<128, 128, 0><<<gGemm128, blk, 0, stream>>>(WV, pWo + l * szQ, bo + l * DIM, Y, M);
    add_ln<<<gLN, blk, 0, stream>>>(X, Y, g1 + l * DIM, be1 + l * DIM, M);

    // FFN on Y (per reference), residual + LN2
    gemm_wmma<128, 256, 1><<<gGemm256, blk, 0, stream>>>(Y, pWf1 + l * szF, bf1 + l * 2 * DIM, T, M);
    gemm_wmma<256, 128, 0><<<gGemm128, blk, 0, stream>>>(T, pWf2 + l * szF, bf2 + l * DIM, V, M);
    add_ln<<<gLN, blk, 0, stream>>>(X, V, g2 + l * DIM, be2 + l * DIM, M);
  }

  // mean over nodes + MLP readout
  hipMemsetAsync(ACC, 0, DIM * sizeof(float), stream);
  colsum<<<512, 128, 0, stream>>>(X, ACC, M);
  readout<<<1, 128, 0, stream>>>(ACC, mW0, mb0, mW1, mb1, mW2, mb2, out, 1.f / (float)M);
}
